// MemoryBank_76940044140845
// MI455X (gfx1250) — compile-verified, matrix operand order checked by reading, not run on hardware
//
#include <hip/hip_runtime.h>

typedef __attribute__((ext_vector_type(16))) __bf16 v16bf;
typedef __attribute__((ext_vector_type(8)))  __bf16 v8bf;
typedef __attribute__((ext_vector_type(4)))  __bf16 v4bf;
typedef __attribute__((ext_vector_type(8)))  float  v8f;

#define DD 256
#define LL 4
#define HH 8
#define HDD 32
#define ROWS 16
#define NTHREADS 256

// bf16 weight workspace layout (element offsets)
#define WQ_OFF   0
#define WK_OFF   (256*256)
#define WV_OFF   (512*256)
#define WO_OFF   (768*256)
#define WF1_OFF  (1024*256)
#define WF2_OFF  (1280*256)
#define WSV_OFF  (1536*256)
#define WTOTAL   (1792*256)

// ---------------------------------------------------------------------------
// fp32 -> bf16 weight conversion (runs once per launch; ~0.5M elements)
// ---------------------------------------------------------------------------
__global__ void cvt_weights_kernel(const float* __restrict__ in_w,
                                   const float* __restrict__ out_w,
                                   const float* __restrict__ f1_w,
                                   const float* __restrict__ f2_w,
                                   const float* __restrict__ sv_w,
                                   __bf16* __restrict__ dst) {
    const int i4 = blockIdx.x * NTHREADS + threadIdx.x;   // one float4 per thread
    if (i4 >= WTOTAL / 4) return;
    const int i = i4 * 4;
    const float* src;
    int off;
    if      (i < WO_OFF)  { src = in_w;  off = i; }
    else if (i < WF1_OFF) { src = out_w; off = i - WO_OFF; }
    else if (i < WF2_OFF) { src = f1_w;  off = i - WF1_OFF; }
    else if (i < WSV_OFF) { src = f2_w;  off = i - WF2_OFF; }
    else                  { src = sv_w;  off = i - WSV_OFF; }
    const float4 v = *(const float4*)(src + off);
    v4bf b;
    b[0] = (__bf16)v.x; b[1] = (__bf16)v.y; b[2] = (__bf16)v.z; b[3] = (__bf16)v.w;
    *(v4bf*)(dst + i) = b;
}

// ---------------------------------------------------------------------------
// GEMM  OUT[MTILES*16][256] = A[MTILES*16][256] @ W^T + bias
// A: LDS bf16 row-major; W: global bf16 row-major [256 out][256 in].
// Per wave: one column tile nt at a time (ni loop NOT unrolled -> one
// 64-VGPR B-fragment set live), B reused across all mt tiles; 2 independent
// accumulator chains interleaved per pass to fill the XDL pipe without
// spilling (live set ~130 VGPRs < 256-VGPR cap at 4 waves/SIMD).
// Fragment layouts per CDNA5 ISA 7.12.2 (16-bit A 16x32; B mirrored; f32 C/D).
// ---------------------------------------------------------------------------
template <int MTILES, bool RELU, bool OUTF32>
__device__ __forceinline__ void wmma_gemm(const __bf16* __restrict__ A,
                                          const __bf16* __restrict__ W,
                                          const float*  __restrict__ bias,
                                          __bf16* __restrict__ outb,
                                          float*  __restrict__ outf,
                                          int tid) {
    const int wave  = tid >> 5;
    const int lane  = tid & 31;
    const int lmod  = lane & 15;
    const int hi    = lane >> 4;           // 0: lanes 0-15, 1: lanes 16-31
    const int khalf = hi * 8;
    constexpr int STEP = (MTILES >= 2) ? 2 : 1;
#pragma unroll 1
    for (int ni = 0; ni < 2; ++ni) {       // 16 column tiles / 8 waves
        const int nt = wave + ni * 8;
        const float bv = bias[nt * 16 + lmod];
        const __bf16* wrow = W + (nt * 16 + lmod) * DD;   // B col n = W row n
        v16bf bfrag[8];
#pragma unroll
        for (int kt = 0; kt < 8; ++kt) {
            const int kb = kt * 32 + khalf;
            v8bf blo = *(const v8bf*)(wrow + kb);
            v8bf bhi = *(const v8bf*)(wrow + kb + 16);
            bfrag[kt] = __builtin_shufflevector(blo, bhi,
                            0,1,2,3,4,5,6,7,8,9,10,11,12,13,14,15);
        }
#pragma unroll 1
        for (int mp = 0; mp < MTILES; mp += STEP) {
            v8f acc0, acc1;
#pragma unroll
            for (int i = 0; i < 8; ++i) { acc0[i] = bv; acc1[i] = bv; }
            const __bf16* arow0 = A + (mp * 16 + lmod) * DD;
            const __bf16* arow1 = A + ((mp + 1) * 16 + lmod) * DD;
#pragma unroll
            for (int kt = 0; kt < 8; ++kt) {              // K = 8 * 32 = 256
                const int kb = kt * 32 + khalf;
                v8bf a0lo = *(const v8bf*)(arow0 + kb);
                v8bf a0hi = *(const v8bf*)(arow0 + kb + 16);
                v16bf af0 = __builtin_shufflevector(a0lo, a0hi,
                                0,1,2,3,4,5,6,7,8,9,10,11,12,13,14,15);
                acc0 = __builtin_amdgcn_wmma_f32_16x16x32_bf16(
                           false, af0, false, bfrag[kt], (short)0, acc0, false, false);
                if (STEP == 2) {
                    v8bf a1lo = *(const v8bf*)(arow1 + kb);
                    v8bf a1hi = *(const v8bf*)(arow1 + kb + 16);
                    v16bf af1 = __builtin_shufflevector(a1lo, a1hi,
                                    0,1,2,3,4,5,6,7,8,9,10,11,12,13,14,15);
                    acc1 = __builtin_amdgcn_wmma_f32_16x16x32_bf16(
                               false, af1, false, bfrag[kt], (short)0, acc1, false, false);
                }
            }
            // D layout: VGPR r -> (M = r + 8*hi, N = lane%16)
            const int col = nt * 16 + lmod;
#pragma unroll
            for (int r = 0; r < 8; ++r) {
                float v0 = acc0[r];
                if (RELU) v0 = v0 > 0.f ? v0 : 0.f;
                const int row0i = mp * 16 + hi * 8 + r;
                if (OUTF32) outf[row0i * DD + col] = v0;
                else        outb[row0i * DD + col] = (__bf16)v0;
                if (STEP == 2) {
                    float v1 = acc1[r];
                    if (RELU) v1 = v1 > 0.f ? v1 : 0.f;
                    const int row1i = (mp + 1) * 16 + hi * 8 + r;
                    if (OUTF32) outf[row1i * DD + col] = v1;
                    else        outb[row1i * DD + col] = (__bf16)v1;
                }
            }
        }
    }
}

// ---------------------------------------------------------------------------
// Fused memory-bank transformer block: 16 tracks per workgroup, 8 waves.
// ---------------------------------------------------------------------------
__global__ __launch_bounds__(NTHREADS, 4)
void memtrack_kernel(const float* __restrict__ xg,    const float* __restrict__ scores,
                     const float* __restrict__ bankg, const int*   __restrict__ mpad,
                     const __bf16* __restrict__ wbf,
                     const float* __restrict__ in_b,  const float* __restrict__ out_b,
                     const float* __restrict__ fc1_b, const float* __restrict__ fc2_b,
                     const float* __restrict__ save_b,
                     const float* __restrict__ g1, const float* __restrict__ b1,
                     const float* __restrict__ g2, const float* __restrict__ b2,
                     float* __restrict__ out_emb, float* __restrict__ out_bank,
                     float* __restrict__ out_mask) {
    __shared__ __bf16 sA[ROWS * DD];          // activation GEMM input (x/ctx/e/new_emb)
    __shared__ __bf16 sB[ROWS * DD];          // ffn hidden
    __shared__ __bf16 sBank[ROWS * LL * DD];  // staged memory bank
    __shared__ __bf16 sK[ROWS * LL * DD];
    __shared__ __bf16 sV[ROWS * LL * DD];
    __shared__ __bf16 sQ[ROWS * DD];
    __shared__ float  sE[ROWS * DD];          // post-LN1 activations (f32)
    __shared__ float  sRes[ROWS * DD];        // f32 GEMM output / residual accum
    __shared__ float  sRedS[NTHREADS];
    __shared__ float  sRedQ[NTHREADS];

    const int tid  = threadIdx.x;
    const int row0 = blockIdx.x * ROWS;

    // ---- stage x and bank as bf16 in LDS (b128 in, b64 out) ----
    {
        const float4* x4 = (const float4*)(xg + (size_t)row0 * DD);
        v4bf* a4 = (v4bf*)sA;
        for (int i = tid; i < ROWS * DD / 4; i += NTHREADS) {
            const float4 v = x4[i];
            v4bf b;
            b[0] = (__bf16)v.x; b[1] = (__bf16)v.y; b[2] = (__bf16)v.z; b[3] = (__bf16)v.w;
            a4[i] = b;
        }
        const float4* bk4 = (const float4*)(bankg + (size_t)row0 * LL * DD);
        v4bf* s4 = (v4bf*)sBank;
        for (int i = tid; i < ROWS * LL * DD / 4; i += NTHREADS) {
            const float4 v = bk4[i];
            v4bf b;
            b[0] = (__bf16)v.x; b[1] = (__bf16)v.y; b[2] = (__bf16)v.z; b[3] = (__bf16)v.w;
            s4[i] = b;
        }
    }
    __syncthreads();

    // ---- Q/K/V projections ----
    wmma_gemm<1,  false, false>(sA,    wbf + WQ_OFF, in_b,          sQ, nullptr, tid);
    wmma_gemm<LL, false, false>(sBank, wbf + WK_OFF, in_b + DD,     sK, nullptr, tid);
    wmma_gemm<LL, false, false>(sBank, wbf + WV_OFF, in_b + 2 * DD, sV, nullptr, tid);
    __syncthreads();

    // ---- tiny attention (H=8 heads, L=4 keys): 1 thread per (row, head) ----
    if (tid < ROWS * HH) {
        const int row  = tid >> 3;
        const int head = tid & 7;
        const int gRow = row0 + row;
        const bool validRow = (mpad[gRow * LL + (LL - 1)] == 0);

        float qf[HDD];
        {
            const v8bf* qp = (const v8bf*)(sQ + row * DD + head * HDD);
#pragma unroll
            for (int c = 0; c < HDD / 8; ++c) {
                const v8bf t = qp[c];
#pragma unroll
                for (int j = 0; j < 8; ++j) qf[c * 8 + j] = (float)t[j];
            }
        }
        float logit[LL];
#pragma unroll
        for (int l = 0; l < LL; ++l) {
            const v8bf* kp = (const v8bf*)(sK + (row * LL + l) * DD + head * HDD);
            float s = 0.f;
#pragma unroll
            for (int c = 0; c < HDD / 8; ++c) {
                const v8bf t = kp[c];
#pragma unroll
                for (int j = 0; j < 8; ++j) s += qf[c * 8 + j] * (float)t[j];
            }
            s *= 0.1767766952966369f;  // 1/sqrt(32)
            const bool pad = validRow && (mpad[gRow * LL + l] != 0);
            logit[l] = pad ? -1e9f : s;
        }
        float m = logit[0];
#pragma unroll
        for (int l = 1; l < LL; ++l) m = fmaxf(m, logit[l]);
        float p[LL], sum = 0.f;
#pragma unroll
        for (int l = 0; l < LL; ++l) { p[l] = __expf(logit[l] - m); sum += p[l]; }
        const float inv = 1.f / sum;

        float ctx[HDD];
#pragma unroll
        for (int i = 0; i < HDD; ++i) ctx[i] = 0.f;
#pragma unroll
        for (int l = 0; l < LL; ++l) {
            const v8bf* vp = (const v8bf*)(sV + (row * LL + l) * DD + head * HDD);
            const float w = p[l] * inv;
#pragma unroll
            for (int c = 0; c < HDD / 8; ++c) {
                const v8bf t = vp[c];
#pragma unroll
                for (int j = 0; j < 8; ++j) ctx[c * 8 + j] += w * (float)t[j];
            }
        }
        v8bf* op = (v8bf*)(sA + row * DD + head * HDD);
#pragma unroll
        for (int c = 0; c < HDD / 8; ++c) {
            v8bf t;
#pragma unroll
            for (int j = 0; j < 8; ++j) t[j] = (__bf16)ctx[c * 8 + j];
            op[c] = t;
        }
    }
    __syncthreads();

    // ---- out_proj -> sRes (f32) ----
    wmma_gemm<1, false, true>(sA, wbf + WO_OFF, out_b, nullptr, sRes, tid);
    __syncthreads();

    // ---- residual + LayerNorm1 -> sE (f32) and sA (bf16) ----
    {
        const int row = tid >> 4, seg = tid & 15;
        const int gRow = row0 + row;
        const float4* r4  = (const float4*)(sRes + row * DD + seg * 16);
        const float4* xr4 = (const float4*)(xg + (size_t)gRow * DD + seg * 16);
        float4 vb[4];
        float s = 0.f, q2 = 0.f;
#pragma unroll
        for (int i = 0; i < 4; ++i) {
            const float4 a = r4[i];
            const float4 b = xr4[i];
            float4 v;
            v.x = a.x + b.x; v.y = a.y + b.y; v.z = a.z + b.z; v.w = a.w + b.w;
            vb[i] = v;
            s  += v.x + v.y + v.z + v.w;
            q2 += v.x * v.x + v.y * v.y + v.z * v.z + v.w * v.w;
        }
        sRedS[tid] = s; sRedQ[tid] = q2;
        __syncthreads();
        float S = 0.f, Q2 = 0.f;
#pragma unroll
        for (int j = 0; j < 16; ++j) { S += sRedS[row * 16 + j]; Q2 += sRedQ[row * 16 + j]; }
        const float mu   = S * (1.f / DD);
        const float var  = Q2 * (1.f / DD) - mu * mu;
        const float rstd = rsqrtf(var + 1e-5f);
        const float4* g4 = (const float4*)(g1 + seg * 16);
        const float4* b4 = (const float4*)(b1 + seg * 16);
        float4* e4 = (float4*)(sE + row * DD + seg * 16);
        v4bf*   a4 = (v4bf*)(sA + row * DD + seg * 16);
#pragma unroll
        for (int i = 0; i < 4; ++i) {
            const float4 g = g4[i], b = b4[i], v = vb[i];
            float4 e;
            e.x = (v.x - mu) * rstd * g.x + b.x;
            e.y = (v.y - mu) * rstd * g.y + b.y;
            e.z = (v.z - mu) * rstd * g.z + b.z;
            e.w = (v.w - mu) * rstd * g.w + b.w;
            e4[i] = e;
            v4bf eb;
            eb[0] = (__bf16)e.x; eb[1] = (__bf16)e.y; eb[2] = (__bf16)e.z; eb[3] = (__bf16)e.w;
            a4[i] = eb;
        }
    }
    __syncthreads();

    // ---- FFN: fc1(+relu) -> sB, fc2 -> sRes ----
    wmma_gemm<1, true,  false>(sA, wbf + WF1_OFF, fc1_b, sB, nullptr, tid);
    __syncthreads();
    wmma_gemm<1, false, true >(sB, wbf + WF2_OFF, fc2_b, nullptr, sRes, tid);
    __syncthreads();

    // ---- residual2 + LayerNorm2 + valid-select, emit new_emb ----
    {
        const int row = tid >> 4, seg = tid & 15;
        const int gRow = row0 + row;
        const float4* e4i = (const float4*)(sE + row * DD + seg * 16);
        const float4* r4  = (const float4*)(sRes + row * DD + seg * 16);
        float4 vb[4];
        float s = 0.f, q2 = 0.f;
#pragma unroll
        for (int i = 0; i < 4; ++i) {
            const float4 a = e4i[i];
            const float4 b = r4[i];
            float4 v;
            v.x = a.x + b.x; v.y = a.y + b.y; v.z = a.z + b.z; v.w = a.w + b.w;
            vb[i] = v;
            s  += v.x + v.y + v.z + v.w;
            q2 += v.x * v.x + v.y * v.y + v.z * v.z + v.w * v.w;
        }
        sRedS[tid] = s; sRedQ[tid] = q2;
        __syncthreads();
        float S = 0.f, Q2 = 0.f;
#pragma unroll
        for (int j = 0; j < 16; ++j) { S += sRedS[row * 16 + j]; Q2 += sRedQ[row * 16 + j]; }
        const float mu   = S * (1.f / DD);
        const float var  = Q2 * (1.f / DD) - mu * mu;
        const float rstd = rsqrtf(var + 1e-5f);
        const bool validRow = (mpad[gRow * LL + (LL - 1)] == 0);
        const float4* g4 = (const float4*)(g2 + seg * 16);
        const float4* b4 = (const float4*)(b2 + seg * 16);
        const float4* x4 = (const float4*)(xg + (size_t)gRow * DD + seg * 16);
        float4* o4 = (float4*)(out_emb + (size_t)gRow * DD + seg * 16);
        v4bf*   a4 = (v4bf*)(sA + row * DD + seg * 16);
#pragma unroll
        for (int i = 0; i < 4; ++i) {
            const float4 g = g4[i], b = b4[i], v = vb[i];
            float4 ne;
            ne.x = (v.x - mu) * rstd * g.x + b.x;
            ne.y = (v.y - mu) * rstd * g.y + b.y;
            ne.z = (v.z - mu) * rstd * g.z + b.z;
            ne.w = (v.w - mu) * rstd * g.w + b.w;
            if (!validRow) ne = x4[i];
            o4[i] = ne;
            v4bf nb;
            nb[0] = (__bf16)ne.x; nb[1] = (__bf16)ne.y; nb[2] = (__bf16)ne.z; nb[3] = (__bf16)ne.w;
            a4[i] = nb;
        }
    }
    __syncthreads();

    // ---- save_proj(new_emb) -> sRes ----
    wmma_gemm<1, false, true>(sA, wbf + WSV_OFF, save_b, nullptr, sRes, tid);
    __syncthreads();

    // ---- bank shift + mask outputs (all b128) ----
    {
        const int row = tid >> 4, seg = tid & 15;
        const int gRow = row0 + row;
        const bool saved = scores[gRow] > 0.f;
        const float4* res4 = (const float4*)(sRes + row * DD);
#pragma unroll
        for (int l = 0; l < LL; ++l) {
#pragma unroll
            for (int i = 0; i < 4; ++i) {
                const int c4 = seg * 4 + i;    // float4 index within a 256-wide row
                float4 v;
                if (saved)
                    v = (l < LL - 1)
                        ? ((const float4*)(bankg + ((size_t)gRow * LL + l + 1) * DD))[c4]
                        : res4[c4];
                else
                    v = ((const float4*)(bankg + ((size_t)gRow * LL + l) * DD))[c4];
                ((float4*)(out_bank + ((size_t)gRow * LL + l) * DD))[c4] = v;
            }
        }
        if (seg == 0) {
#pragma unroll
            for (int l = 0; l < LL; ++l) {
                float mv;
                if (saved) mv = (l < LL - 1) ? (float)(mpad[gRow * LL + l + 1] != 0) : 0.f;
                else       mv = (float)(mpad[gRow * LL + l] != 0);
                out_mask[gRow * LL + l] = mv;
            }
        }
    }
}

// ---------------------------------------------------------------------------
extern "C" void kernel_launch(void* const* d_in, const int* in_sizes, int n_in,
                              void* d_out, int out_size, void* d_ws, size_t ws_size,
                              hipStream_t stream) {
    const float* xg     = (const float*)d_in[0];
    const float* scores = (const float*)d_in[1];
    const float* bankg  = (const float*)d_in[2];
    const int*   mpad   = (const int*)d_in[3];
    const float* sv_w   = (const float*)d_in[4];
    const float* sv_b   = (const float*)d_in[5];
    const float* in_w   = (const float*)d_in[6];
    const float* in_b   = (const float*)d_in[7];
    const float* out_w  = (const float*)d_in[8];
    const float* out_b  = (const float*)d_in[9];
    const float* f1_w   = (const float*)d_in[10];
    const float* f1_b   = (const float*)d_in[11];
    const float* f2_w   = (const float*)d_in[12];
    const float* f2_b   = (const float*)d_in[13];
    const float* g1     = (const float*)d_in[14];
    const float* b1     = (const float*)d_in[15];
    const float* g2     = (const float*)d_in[16];
    const float* b2     = (const float*)d_in[17];

    const int N = in_sizes[1];  // scores is [N]

    __bf16* wbf = (__bf16*)d_ws;

    float* out_emb  = (float*)d_out;
    float* out_bank = out_emb + (size_t)N * DD;
    float* out_mask = out_bank + (size_t)N * LL * DD;

    cvt_weights_kernel<<<(WTOTAL / 4 + NTHREADS - 1) / NTHREADS, NTHREADS, 0, stream>>>(
        in_w, out_w, f1_w, f2_w, sv_w, wbf);

    memtrack_kernel<<<N / ROWS, NTHREADS, 0, stream>>>(
        xg, scores, bankg, mpad, wbf,
        in_b, out_b, f1_b, f2_b, sv_b,
        g1, b1, g2, b2,
        out_emb, out_bank, out_mask);
}
